// GraphTransformer_33500744909055
// MI455X (gfx1250) — matrix operand order repaired and checked by reading, not compiled, not run on hardware
//
#include <hip/hip_runtime.h>
#include <hip/hip_bf16.h>

// ---------------- WMMA vector types (gfx1250, wave32) ----------------
typedef __attribute__((ext_vector_type(16))) _Float16 v16h;
typedef __attribute__((ext_vector_type(8)))  _Float16 v8h;
typedef __attribute__((ext_vector_type(8)))  float    v8f;

union V16H { v16h v; v8h h[2]; };

#define BB 64
#define NN 256
#define TT 10240
#define DIM 128
#define HH 9
#define DHH 8
#define RR 8
#define FF 512
#define MM (BB * NN)          // 16384 dense rows
#define HD 72                 // H*DH
#define HDPAD 96              // padded K for Wo GEMM (multiple of 32)
#define QKVPAD 80             // padded N for QKV GEMM (multiple of 16)

#define WAVES_PER_BLOCK 4

// ---------------- helpers: dense-batch construction ----------------
__global__ void counts_kernel(const int* __restrict__ batch, int* __restrict__ counts) {
    int t = blockIdx.x * blockDim.x + threadIdx.x;
    if (t < TT) atomicAdd(&counts[batch[t]], 1);
}

__global__ void starts_kernel(const int* __restrict__ counts, int* __restrict__ starts) {
    if (threadIdx.x == 0 && blockIdx.x == 0) {
        int acc = 0;
        for (int b = 0; b < BB; ++b) { starts[b] = acc; acc += counts[b]; }
    }
}

__global__ void scatter_kernel(const int* __restrict__ x, const float* __restrict__ pos,
                               const int* __restrict__ batch, const float* __restrict__ embed,
                               const int* __restrict__ starts,
                               float* __restrict__ hdense, float* __restrict__ coors,
                               float* __restrict__ maskf) {
    int t = blockIdx.x;          // token
    int d = threadIdx.x;         // 128 feature lanes
    int b = batch[t];
    int idx = t - starts[b];
    if (idx >= NN) return;       // uniform per block
    int row = b * NN + idx;
    hdense[(size_t)row * DIM + d] = embed[(size_t)x[t] * DIM + d];
    if (d < 3) coors[(size_t)row * 3 + d] = pos[(size_t)t * 3 + d];
    if (d == 0) maskf[row] = 1.0f;
}

// ---------------- layer norm (128 features, 128 threads/row) ----------------
__global__ void ln_kernel(const float* __restrict__ h, const float* __restrict__ g,
                          const float* __restrict__ beta, _Float16* __restrict__ out) {
    int row = blockIdx.x;
    int d = threadIdx.x;
    float xv = h[(size_t)row * DIM + d];
    __shared__ float sm[DIM];
    sm[d] = xv;
    __syncthreads();
    for (int s = 64; s > 0; s >>= 1) { if (d < s) sm[d] += sm[d + s]; __syncthreads(); }
    float mu = sm[0] * (1.0f / DIM);
    __syncthreads();
    float xm = xv - mu;
    sm[d] = xm * xm;
    __syncthreads();
    for (int s = 64; s > 0; s >>= 1) { if (d < s) sm[d] += sm[d + s]; __syncthreads(); }
    float var = sm[0] * (1.0f / DIM);
    float y = xm * rsqrtf(var + 1e-5f) * g[d] + beta[d];
    out[(size_t)row * DIM + d] = (_Float16)y;
}

// ---------------- weight f32 -> f16 transpose with zero padding ----------------
// W: [K, Ncol] row-major f32   ->  WT: [NPad, KPad] row-major f16 (zero padded)
__global__ void transpose_f16(const float* __restrict__ W, _Float16* __restrict__ WT,
                              int K, int Ncol, int KPad, int NPad) {
    int i = blockIdx.x * blockDim.x + threadIdx.x;
    if (i >= NPad * KPad) return;
    int n = i / KPad, k = i % KPad;
    float v = (n < Ncol && k < K) ? W[(size_t)k * Ncol + n] : 0.0f;
    WT[(size_t)n * KPad + k] = (_Float16)v;
}

// ---------------- multi-tile WMMA GEMM: 16 x (16*NT) output per wave ----------------
// A  : [M, KA] f16 row-major, KA multiple of 32
// BT : [NPad, KA] f16 row-major (B column-major; K contiguous per output column)
// One A tile load per k-step feeds NT WMMAs (A reuse across the NT column tiles).
// Epilogue modes:
//  0: outf[row*Nreal + col]  = D                            (QKV)
//  1: hres[row*DIM + col]   += D                            (attn out proj, residual)
//  2: out16[row*NPad + col]  = f16(gelu(D + bias[col]))     (FF1)
//  3: outf[row*DIM + col]    = (hres + D + bias[col]) * maskf[row]   (FF2+mask)
template <int NT>
__global__ void gemm_wmma(const _Float16* __restrict__ A, const _Float16* __restrict__ BT,
                          int KA, int NPad, int Nreal, int mode,
                          float* __restrict__ outf, float* __restrict__ hres,
                          const float* __restrict__ bias, _Float16* __restrict__ out16,
                          const float* __restrict__ maskf) {
    int lane = threadIdx.x & 31;
    int wave = blockIdx.x * WAVES_PER_BLOCK + (threadIdx.x >> 5);
    int ngroups = NPad / (16 * NT);          // column-tile groups per row band
    int tm = wave / ngroups;
    int tg = wave % ngroups;                 // group of NT column tiles
    int r  = lane & 15;
    int hh = lane >> 4;

    // A lane layout (16-bit A 16x32): halves[0..7]=A[r][hh*8+0..7], halves[8..15]=A[r][16+hh*8+0..7]
    const _Float16* arow = A + (size_t)(tm * 16 + r) * KA + hh * 8;
    // B lane layout (16-bit B 32x16): halves[0..7]=B[hh*8+0..7][col], halves[8..15]=B[16+hh*8+0..7][col]
    const _Float16* brow = BT + (size_t)(tg * NT * 16 + r) * KA + hh * 8;
    const size_t bstride = (size_t)16 * KA;  // halfs between adjacent column tiles

    v8f acc[NT];
    #pragma unroll
    for (int t = 0; t < NT; ++t) acc[t] = (v8f){};

    for (int k0 = 0; k0 < KA; k0 += 32) {
        V16H a;
        a.h[0] = *(const v8h*)(arow + k0);
        a.h[1] = *(const v8h*)(arow + k0 + 16);
        if (k0 + 32 < KA) __builtin_prefetch(arow + k0 + 32, 0, 3);   // global_prefetch_b8
        #pragma unroll
        for (int t = 0; t < NT; ++t) {
            V16H b;
            const _Float16* bp = brow + (size_t)t * bstride + k0;
            b.h[0] = *(const v8h*)(bp);
            b.h[1] = *(const v8h*)(bp + 16);
            acc[t] = __builtin_amdgcn_wmma_f32_16x16x32_f16(false, a.v, false, b.v,
                                                            (short)0, acc[t], false, false);
        }
    }

    int row0 = tm * 16 + hh * 8;
    #pragma unroll
    for (int t = 0; t < NT; ++t) {
        int col = (tg * NT + t) * 16 + r;
        if (col >= Nreal) continue;
        if (mode == 0) {
            #pragma unroll
            for (int v = 0; v < 8; ++v)
                outf[(size_t)(row0 + v) * Nreal + col] = acc[t][v];
        } else if (mode == 1) {
            #pragma unroll
            for (int v = 0; v < 8; ++v)
                hres[(size_t)(row0 + v) * DIM + col] += acc[t][v];
        } else if (mode == 2) {
            float bc = bias[col];
            #pragma unroll
            for (int v = 0; v < 8; ++v) {
                float xx = acc[t][v] + bc;
                float gl = 0.5f * xx *
                           (1.0f + tanhf(0.7978845608f * (xx + 0.044715f * xx * xx * xx)));
                out16[(size_t)(row0 + v) * NPad + col] = (_Float16)gl;
            }
        } else {
            float bc = bias[col];
            #pragma unroll
            for (int v = 0; v < 8; ++v) {
                int row = row0 + v;
                outf[(size_t)row * DIM + col] =
                    (hres[(size_t)row * DIM + col] + acc[t][v] + bc) * maskf[row];
            }
        }
    }
}

// ---------------- scalar flash attention with RBF distance bias ----------------
// one thread per (b, i, h); online softmax over j = 0..N-1
__global__ void attn_kernel(const float* __restrict__ q, const float* __restrict__ k,
                            const float* __restrict__ v, const float* __restrict__ coors,
                            const float* __restrict__ maskf, const float* __restrict__ Wrbf,
                            _Float16* __restrict__ o_pad) {
    int gid = blockIdx.x * blockDim.x + threadIdx.x;
    if (gid >= MM * HH) return;
    int hI = gid % HH;
    int row = gid / HH;           // b*N + i
    int base = row & ~(NN - 1);   // b*N

    float qv[DHH];
    #pragma unroll
    for (int d = 0; d < DHH; ++d)
        qv[d] = q[(size_t)row * HD + hI * DHH + d] * 0.35355339059327373f; // 1/sqrt(DH)
    float wr[RR];
    #pragma unroll
    for (int r = 0; r < RR; ++r) wr[r] = Wrbf[r * HH + hI];

    float cx = coors[(size_t)row * 3 + 0];
    float cy = coors[(size_t)row * 3 + 1];
    float cz = coors[(size_t)row * 3 + 2];
    float mi = maskf[row];

    float mmax = -1e30f, ssum = 0.0f;
    float acc[DHH];
    #pragma unroll
    for (int d = 0; d < DHH; ++d) acc[d] = 0.0f;

    const float cstep = 8.0f / 7.0f;  // linspace(0, 8, 8)
    for (int j = 0; j < NN; ++j) {
        int rj = base + j;
        float logit = -1e30f;
        if (mi != 0.0f && maskf[rj] != 0.0f) {
            const float* kj = k + (size_t)rj * HD + hI * DHH;
            float dot = 0.0f;
            #pragma unroll
            for (int d = 0; d < DHH; ++d) dot += qv[d] * kj[d];
            float dx = cx - coors[(size_t)rj * 3 + 0];
            float dy = cy - coors[(size_t)rj * 3 + 1];
            float dz = cz - coors[(size_t)rj * 3 + 2];
            float dist = sqrtf(dx * dx + dy * dy + dz * dz + 1e-8f);
            float bsum = 0.0f;
            #pragma unroll
            for (int r = 0; r < RR; ++r) {
                float t = dist - (float)r * cstep;
                bsum += wr[r] * __expf(-0.5f * t * t);
            }
            logit = dot + bsum;
        }
        float newm = fmaxf(mmax, logit);
        float scale = __expf(mmax - newm);
        float e = __expf(logit - newm);
        ssum = ssum * scale + e;
        const float* vj = v + (size_t)rj * HD + hI * DHH;
        #pragma unroll
        for (int d = 0; d < DHH; ++d) acc[d] = acc[d] * scale + e * vj[d];
        mmax = newm;
    }
    float inv = 1.0f / ssum;
    #pragma unroll
    for (int d = 0; d < DHH; ++d)
        o_pad[(size_t)row * HDPAD + hI * DHH + d] = (_Float16)(acc[d] * inv);
}

// ---------------- launcher ----------------
extern "C" void kernel_launch(void* const* d_in, const int* in_sizes, int n_in,
                              void* d_out, int out_size, void* d_ws, size_t ws_size,
                              hipStream_t stream) {
    const int*   x     = (const int*)d_in[0];
    const float* pos   = (const float*)d_in[1];
    const int*   batch = (const int*)d_in[2];
    const float* embed = (const float*)d_in[3];
    const float* Wq    = (const float*)d_in[4];
    const float* Wk    = (const float*)d_in[5];
    const float* Wv    = (const float*)d_in[6];
    const float* Wrbf  = (const float*)d_in[7];
    const float* Wo    = (const float*)d_in[8];
    const float* ln1_g = (const float*)d_in[9];
    const float* ln1_b = (const float*)d_in[10];
    const float* ln2_g = (const float*)d_in[11];
    const float* ln2_b = (const float*)d_in[12];
    const float* W1    = (const float*)d_in[13];
    const float* b1    = (const float*)d_in[14];
    const float* W2    = (const float*)d_in[15];
    const float* b2    = (const float*)d_in[16];
    float* out = (float*)d_out;

    char* ws = (char*)d_ws;
    size_t off = 0;
    auto alloc = [&](size_t bytes) -> char* {
        char* p = ws + off;
        off = (off + bytes + 255) & ~(size_t)255;
        return p;
    };

    int*       counts = (int*)alloc(BB * 4);
    int*       starts = (int*)alloc(BB * 4);
    float*     maskf  = (float*)alloc((size_t)MM * 4);
    float*     hdense = (float*)alloc((size_t)MM * DIM * 4);
    float*     coors  = (float*)alloc((size_t)MM * 3 * 4);
    _Float16*  hn     = (_Float16*)alloc((size_t)MM * DIM * 2);   // reused for LN1 and LN2
    float*     qbuf   = (float*)alloc((size_t)MM * HD * 4);
    float*     kbuf   = (float*)alloc((size_t)MM * HD * 4);
    float*     vbuf   = (float*)alloc((size_t)MM * HD * 4);
    _Float16*  o_pad  = (_Float16*)alloc((size_t)MM * HDPAD * 2);
    _Float16*  ff     = (_Float16*)alloc((size_t)MM * FF * 2);
    _Float16*  WqT    = (_Float16*)alloc((size_t)QKVPAD * DIM * 2);
    _Float16*  WkT    = (_Float16*)alloc((size_t)QKVPAD * DIM * 2);
    _Float16*  WvT    = (_Float16*)alloc((size_t)QKVPAD * DIM * 2);
    _Float16*  WoT    = (_Float16*)alloc((size_t)DIM * HDPAD * 2);
    _Float16*  W1T    = (_Float16*)alloc((size_t)FF * DIM * 2);
    _Float16*  W2T    = (_Float16*)alloc((size_t)DIM * FF * 2);
    (void)ws_size; (void)in_sizes; (void)n_in; (void)out_size;

    // re-initialize scratch that must start at zero (capture-safe async memsets)
    hipMemsetAsync(counts, 0, BB * 4, stream);
    hipMemsetAsync(maskf,  0, (size_t)MM * 4, stream);
    hipMemsetAsync(hdense, 0, (size_t)MM * DIM * 4, stream);
    hipMemsetAsync(coors,  0, (size_t)MM * 3 * 4, stream);
    hipMemsetAsync(o_pad,  0, (size_t)MM * HDPAD * 2, stream);

    // dense batch construction
    counts_kernel<<<(TT + 255) / 256, 256, 0, stream>>>(batch, counts);
    starts_kernel<<<1, 32, 0, stream>>>(counts, starts);
    scatter_kernel<<<TT, DIM, 0, stream>>>(x, pos, batch, embed, starts, hdense, coors, maskf);

    // weight transposes / f16 conversion
    transpose_f16<<<(QKVPAD * DIM + 255) / 256, 256, 0, stream>>>(Wq, WqT, DIM, HD, DIM, QKVPAD);
    transpose_f16<<<(QKVPAD * DIM + 255) / 256, 256, 0, stream>>>(Wk, WkT, DIM, HD, DIM, QKVPAD);
    transpose_f16<<<(QKVPAD * DIM + 255) / 256, 256, 0, stream>>>(Wv, WvT, DIM, HD, DIM, QKVPAD);
    transpose_f16<<<(DIM * HDPAD + 255) / 256, 256, 0, stream>>>(Wo, WoT, HD, DIM, HDPAD, DIM);
    transpose_f16<<<(FF * DIM + 255) / 256, 256, 0, stream>>>(W1, W1T, DIM, FF, DIM, FF);
    transpose_f16<<<(DIM * FF + 255) / 256, 256, 0, stream>>>(W2, W2T, FF, DIM, FF, DIM);

    // LN1 -> f16
    ln_kernel<<<MM, DIM, 0, stream>>>(hdense, ln1_g, ln1_b, hn);

    // QKV projections via WMMA (M=16384, K=128, Npad=80: one 5-tile group per row band)
    {
        int waves = (MM / 16) * (QKVPAD / 80);               // 1024 waves, NT=5
        int blocks = waves / WAVES_PER_BLOCK;
        gemm_wmma<5><<<blocks, 32 * WAVES_PER_BLOCK, 0, stream>>>(
            hn, WqT, DIM, QKVPAD, HD, 0, qbuf, nullptr, nullptr, nullptr, nullptr);
        gemm_wmma<5><<<blocks, 32 * WAVES_PER_BLOCK, 0, stream>>>(
            hn, WkT, DIM, QKVPAD, HD, 0, kbuf, nullptr, nullptr, nullptr, nullptr);
        gemm_wmma<5><<<blocks, 32 * WAVES_PER_BLOCK, 0, stream>>>(
            hn, WvT, DIM, QKVPAD, HD, 0, vbuf, nullptr, nullptr, nullptr, nullptr);
    }

    // attention with RBF bias (scalar flash attention, K=DH=8)
    attn_kernel<<<(MM * HH + 255) / 256, 256, 0, stream>>>(qbuf, kbuf, vbuf, coors, maskf, Wrbf, o_pad);

    // output projection + residual: h += o @ Wo   (K=96, N=128, NT=4 -> 2 groups)
    {
        int waves = (MM / 16) * (DIM / 64);
        gemm_wmma<4><<<waves / WAVES_PER_BLOCK, 32 * WAVES_PER_BLOCK, 0, stream>>>(
            o_pad, WoT, HDPAD, DIM, DIM, 1, nullptr, hdense, nullptr, nullptr, nullptr);
    }

    // LN2 -> f16 (reuse hn)
    ln_kernel<<<MM, DIM, 0, stream>>>(hdense, ln2_g, ln2_b, hn);

    // FF1: gelu(hn @ W1 + b1) -> f16 (K=128, N=512, NT=4 -> 8 groups)
    {
        int waves = (MM / 16) * (FF / 64);
        gemm_wmma<4><<<waves / WAVES_PER_BLOCK, 32 * WAVES_PER_BLOCK, 0, stream>>>(
            hn, W1T, DIM, FF, FF, 2, nullptr, nullptr, b1, ff, nullptr);
    }

    // FF2: out = (h + ff @ W2 + b2) * mask   (K=512, N=128, NT=4 -> 2 groups)
    {
        int waves = (MM / 16) * (DIM / 64);
        gemm_wmma<4><<<waves / WAVES_PER_BLOCK, 32 * WAVES_PER_BLOCK, 0, stream>>>(
            ff, W2T, FF, DIM, DIM, 3, out, hdense, b2, nullptr, maskf);
    }
}
